// NeuralODEWarpV2_36215164240625
// MI455X (gfx1250) — matrix operand order repaired and checked by reading, not compiled, not run on hardware
//
#include <hip/hip_runtime.h>

typedef __attribute__((ext_vector_type(16))) _Float16 v16h;
typedef __attribute__((ext_vector_type(8)))  _Float16 v8h;
typedef __attribute__((ext_vector_type(8)))  float    v8f;

#define OMEGA0f 30.0f
#define HSTEP   (1.0f/16.0f)
#define NWAVES  8
#define BLOCKT  256

// ---------------------------------------------------------------------------
// Kernel 1: global min/max of t column -> t_ref = (min+max)/2  (EPS cancels)
// ---------------------------------------------------------------------------
__global__ void tref_reduce_kernel(const float* __restrict__ txy,
                                   float* __restrict__ ws, int n) {
  __shared__ float smn[BLOCKT];
  __shared__ float smx[BLOCKT];
  float mn = 3.0e38f, mx = -3.0e38f;
  for (int i = threadIdx.x; i < n; i += BLOCKT) {
    float t = txy[3 * i];
    mn = fminf(mn, t);
    mx = fmaxf(mx, t);
  }
  smn[threadIdx.x] = mn;
  smx[threadIdx.x] = mx;
  __syncthreads();
  for (int s = BLOCKT / 2; s > 0; s >>= 1) {
    if (threadIdx.x < s) {
      smn[threadIdx.x] = fminf(smn[threadIdx.x], smn[threadIdx.x + s]);
      smx[threadIdx.x] = fmaxf(smx[threadIdx.x], smx[threadIdx.x + s]);
    }
    __syncthreads();
  }
  if (threadIdx.x == 0) ws[0] = 0.5f * (smn[0] + smx[0]);
}

// ---------------------------------------------------------------------------
// Kernel 2: neural-ODE DOPRI5 integration, hidden layers via WMMA f16->f32
// One wave = 16 points; block = 8 waves = 128 points.
// ---------------------------------------------------------------------------
__global__ __launch_bounds__(BLOCKT, 1)
void node_warp_kernel(const float* __restrict__ txy,
                      const float* __restrict__ W0, const float* __restrict__ b0,
                      const float* __restrict__ W1, const float* __restrict__ b1,
                      const float* __restrict__ W2, const float* __restrict__ b2,
                      const float* __restrict__ Wo, const float* __restrict__ bo,
                      const float* __restrict__ ws, float* __restrict__ out, int n)
{
  // packed B-fragment weights (f16), lane-contiguous: frag=(nt*4+kc), then
  // [lane][0..15] halves per lane.  lane<16: col=nt*16+lane, K=32kc+(i<8?i:i+8) ;
  // lane>=16: col=nt*16+lane-16,   K=32kc+8+(i<8?i:i+8)
  __shared__ _Float16 w1p[16384];              // 32 KB
  __shared__ _Float16 w2p[16384];              // 32 KB
  __shared__ _Float16 wop[2048];               //  4 KB (Wo padded 128x16)
  __shared__ float    w0s[384];                // W0 (3x128) f32
  __shared__ float    b0s[128], b1s[128], b2s[128];
  __shared__ float    bos[16];
  __shared__ _Float16 hbuf[NWAVES][2048];      // per-wave 16x128 f16 activations
  __shared__ float    fbuf[NWAVES][256];       // per-wave 16x16 f32 out tile
  __shared__ float    sbuf[NWAVES][16][4];     // per-wave stage coords (t,x,y)

  const int tid = threadIdx.x;

  // ---- one-time pack of weights into LDS ----
  for (int idx = tid; idx < 16384; idx += BLOCKT) {
    int i    = idx & 15;
    int lane = (idx >> 4) & 31;
    int frag = idx >> 9;                 // nt*4 + kc
    int kc   = frag & 3;
    int nt   = frag >> 2;
    int col  = nt * 16 + (lane & 15);
    int k    = kc * 32 + ((lane & 16) ? 8 : 0) + (i < 8 ? i : i + 8);
    w1p[idx] = (_Float16)W1[k * 128 + col];
    w2p[idx] = (_Float16)W2[k * 128 + col];
  }
  for (int idx = tid; idx < 2048; idx += BLOCKT) {
    int i    = idx & 15;
    int lane = (idx >> 4) & 31;
    int kc   = idx >> 9;
    int col  = lane & 15;
    int k    = kc * 32 + ((lane & 16) ? 8 : 0) + (i < 8 ? i : i + 8);
    wop[idx] = (col < 2) ? (_Float16)Wo[k * 2 + col] : (_Float16)0.0f;
  }
  for (int idx = tid; idx < 384; idx += BLOCKT) w0s[idx] = W0[idx];
  if (tid < 128) { b0s[tid] = b0[tid]; b1s[tid] = b1[tid]; b2s[tid] = b2[tid]; }
  if (tid < 16)  bos[tid] = (tid < 2) ? bo[tid] : 0.0f;
  __syncthreads();

  const int wave = tid >> 5;
  const int lane = tid & 31;
  const float tref = ws[0];

  const int gidx = blockIdx.x * 128 + wave * 16 + (lane & 15);
  int pt = gidx;
  if (pt > n - 1) pt = n - 1;
  float t0 = txy[3 * pt + 0];
  float x  = txy[3 * pt + 1];
  float y  = txy[3 * pt + 2];
  float dt = tref - t0;

  // WMMA fragment addressing for this lane
  const int arow  = lane & 15;                 // A: matrix row
  const int aoff  = (lane & 16) ? 8 : 0;       // A: K phase
  const int ccol  = lane & 15;                 // C/D: column
  const int crow0 = (lane & 16) ? 8 : 0;       // C/D: row base

  // ---- hidden layer: h = sin(h @ W + b), 16x128 per wave, in-place ----
  auto hidden = [&](const _Float16* Bp, const float* bias) {
    v16h a[4];
#pragma unroll
    for (int kc = 0; kc < 4; ++kc) {
      const _Float16* p = &hbuf[wave][arow * 128 + kc * 32 + aoff];
      v8h lo = *(const v8h*)p;
      v8h hi = *(const v8h*)(p + 16);
      a[kc] = __builtin_shufflevector(lo, hi, 0,1,2,3,4,5,6,7,8,9,10,11,12,13,14,15);
    }
#pragma unroll
    for (int nt = 0; nt < 8; ++nt) {
      float bv = bias[nt * 16 + ccol];
      v8f acc = {bv, bv, bv, bv, bv, bv, bv, bv};
#pragma unroll
      for (int kc = 0; kc < 4; ++kc) {
        v16h b = *(const v16h*)&Bp[((nt * 4 + kc) * 32 + lane) * 16];
        acc = __builtin_amdgcn_wmma_f32_16x16x32_f16(false, a[kc], false, b,
                                                     (short)0, acc, false, false);
      }
      int col = nt * 16 + ccol;
#pragma unroll
      for (int r = 0; r < 8; ++r)
        hbuf[wave][(crow0 + r) * 128 + col] = (_Float16)__sinf(acc[r]);
    }
  };

  // ---- output layer: flow = h @ Wo + bo (Wo zero-padded to 128x16) ----
  auto outlayer = [&](float& f0, float& f1) {
    v16h a[4];
#pragma unroll
    for (int kc = 0; kc < 4; ++kc) {
      const _Float16* p = &hbuf[wave][arow * 128 + kc * 32 + aoff];
      v8h lo = *(const v8h*)p;
      v8h hi = *(const v8h*)(p + 16);
      a[kc] = __builtin_shufflevector(lo, hi, 0,1,2,3,4,5,6,7,8,9,10,11,12,13,14,15);
    }
    float bv = bos[ccol];
    v8f acc = {bv, bv, bv, bv, bv, bv, bv, bv};
#pragma unroll
    for (int kc = 0; kc < 4; ++kc) {
      v16h b = *(const v16h*)&wop[(kc * 32 + lane) * 16];
      acc = __builtin_amdgcn_wmma_f32_16x16x32_f16(false, a[kc], false, b,
                                                   (short)0, acc, false, false);
    }
#pragma unroll
    for (int r = 0; r < 8; ++r)
      fbuf[wave][(crow0 + r) * 16 + ccol] = acc[r];
    f0 = fbuf[wave][(lane & 15) * 16 + 0];
    f1 = fbuf[wave][(lane & 15) * 16 + 1];
  };

  // ---- one ODE-func evaluation (flow at stage coords) ----
  auto eval = [&](float tnorm, float xs, float ys, float& f0, float& f1) {
    float treal = t0 + tnorm * dt;
    if (lane < 16) {
      sbuf[wave][lane][0] = treal;
      sbuf[wave][lane][1] = xs;
      sbuf[wave][lane][2] = ys;
    }
    // layer 0: h0 = sin(omega0 * (coords @ W0 + b0)),  K=3 -> VALU
    for (int m = 0; m < 16; ++m) {
      float tt = sbuf[wave][m][0];
      float xx = sbuf[wave][m][1];
      float yy = sbuf[wave][m][2];
#pragma unroll
      for (int jj = 0; jj < 4; ++jj) {
        int j = lane + 32 * jj;
        float pre = OMEGA0f * (tt * w0s[j] + xx * w0s[128 + j] + yy * w0s[256 + j] + b0s[j]);
        hbuf[wave][m * 128 + j] = (_Float16)__sinf(pre);
      }
    }
    hidden(w1p, b1s);
    hidden(w2p, b2s);
    outlayer(f0, f1);
  };

  // ---- DOPRI5, 16 fixed steps, stages unrolled ----
  for (int step = 0; step < 16; ++step) {
    float tn = (float)step * HSTEP;
    float f0, f1;
    float k0x, k0y, k1x, k1y, k2x, k2y, k3x, k3y, k4x, k4y, k5x, k5y;

    eval(tn, x, y, f0, f1);
    k0x = f0 * dt; k0y = f1 * dt;

    eval(tn + 0.2f * HSTEP,
         x + HSTEP * (0.2f * k0x),
         y + HSTEP * (0.2f * k0y), f0, f1);
    k1x = f0 * dt; k1y = f1 * dt;

    eval(tn + 0.3f * HSTEP,
         x + HSTEP * ((3.0f/40.0f) * k0x + (9.0f/40.0f) * k1x),
         y + HSTEP * ((3.0f/40.0f) * k0y + (9.0f/40.0f) * k1y), f0, f1);
    k2x = f0 * dt; k2y = f1 * dt;

    eval(tn + 0.8f * HSTEP,
         x + HSTEP * ((44.0f/45.0f) * k0x + (-56.0f/15.0f) * k1x + (32.0f/9.0f) * k2x),
         y + HSTEP * ((44.0f/45.0f) * k0y + (-56.0f/15.0f) * k1y + (32.0f/9.0f) * k2y), f0, f1);
    k3x = f0 * dt; k3y = f1 * dt;

    eval(tn + (8.0f/9.0f) * HSTEP,
         x + HSTEP * ((19372.0f/6561.0f) * k0x + (-25360.0f/2187.0f) * k1x +
                      (64448.0f/6561.0f) * k2x + (-212.0f/729.0f) * k3x),
         y + HSTEP * ((19372.0f/6561.0f) * k0y + (-25360.0f/2187.0f) * k1y +
                      (64448.0f/6561.0f) * k2y + (-212.0f/729.0f) * k3y), f0, f1);
    k4x = f0 * dt; k4y = f1 * dt;

    eval(tn + HSTEP,
         x + HSTEP * ((9017.0f/3168.0f) * k0x + (-355.0f/33.0f) * k1x +
                      (46732.0f/5247.0f) * k2x + (49.0f/176.0f) * k3x +
                      (-5103.0f/18656.0f) * k4x),
         y + HSTEP * ((9017.0f/3168.0f) * k0y + (-355.0f/33.0f) * k1y +
                      (46732.0f/5247.0f) * k2y + (49.0f/176.0f) * k3y +
                      (-5103.0f/18656.0f) * k4y), f0, f1);
    k5x = f0 * dt; k5y = f1 * dt;

    x += HSTEP * ((35.0f/384.0f) * k0x + (500.0f/1113.0f) * k2x + (125.0f/192.0f) * k3x +
                  (-2187.0f/6784.0f) * k4x + (11.0f/84.0f) * k5x);
    y += HSTEP * ((35.0f/384.0f) * k0y + (500.0f/1113.0f) * k2y + (125.0f/192.0f) * k3y +
                  (-2187.0f/6784.0f) * k4y + (11.0f/84.0f) * k5y);
  }

  if (lane < 16 && gidx < n) {
    out[3 * gidx + 0] = tref;
    out[3 * gidx + 1] = x;
    out[3 * gidx + 2] = y;
  }
}

// ---------------------------------------------------------------------------
extern "C" void kernel_launch(void* const* d_in, const int* in_sizes, int n_in,
                              void* d_out, int out_size, void* d_ws, size_t ws_size,
                              hipStream_t stream) {
  const float* txy = (const float*)d_in[0];
  const float* W0  = (const float*)d_in[1];
  const float* b0  = (const float*)d_in[2];
  const float* W1  = (const float*)d_in[3];
  const float* b1  = (const float*)d_in[4];
  const float* W2  = (const float*)d_in[5];
  const float* b2  = (const float*)d_in[6];
  const float* Wo  = (const float*)d_in[7];
  const float* bo  = (const float*)d_in[8];
  (void)n_in; (void)out_size; (void)ws_size;

  int n = in_sizes[0] / 3;
  float* ws  = (float*)d_ws;
  float* out = (float*)d_out;

  tref_reduce_kernel<<<1, BLOCKT, 0, stream>>>(txy, ws, n);
  int nb = (n + 127) / 128;
  node_warp_kernel<<<nb, BLOCKT, 0, stream>>>(txy, W0, b0, W1, b1, W2, b2,
                                              Wo, bo, ws, out, n);
}
